// MoELayer_5540507812069
// MI455X (gfx1250) — matrix operand order repaired and checked by reading, not compiled, not run on hardware
//
#include <hip/hip_runtime.h>
#include <math.h>
#include <stdint.h>

// ---------------- problem constants (match reference) ----------------
#define NTOK   8192      // B*T
#define CDIM   1024      // C
#define HDIM   4096      // H
#define NEXP   8         // E
#define TM     64        // tokens per tile
#define HBF    64        // H chunk width (bf16/LDS-panel path)
#define HB32   128       // H chunk width (f32 fallback path)
#define CS     1040      // Xs row stride (halves), 16B aligned, padded
#define HSF    72        // Hs row stride (halves), bf16 path
#define HS32   144       // Hs row stride (halves), f32 path
#define PS_UP  72        // up panel row stride  [1024][72]
#define MAXT_GRID 264    // max tiles: 8 + 16384/64
#define TILE_TAB  512    // table capacity

typedef __attribute__((ext_vector_type(16))) __bf16 v16bf;
typedef __attribute__((ext_vector_type(8)))  float  v8f;

union FragBF {
    v16bf v;
    uint4 u[2];
    __bf16 e[16];
};

__device__ __forceinline__ unsigned short bfbits(float f) {
    union { __bf16 b; unsigned short u; } cv;
    cv.b = (__bf16)f;
    return cv.u;
}

__device__ __forceinline__ float gelu_exact(float v) {
    return 0.5f * v * (1.0f + erff(v * 0.70710678118654752f));
}

// ---- CDNA5 async global->LDS copy (ASYNCcnt) ----
__device__ __forceinline__ void async_copy16(const void* g, void* l) {
    unsigned lds = (unsigned)(uintptr_t)l;   // generic shared ptr: low 32 bits = LDS addr
    asm volatile("global_load_async_to_lds_b128 %0, %1, off"
                 :: "v"(lds), "v"(g) : "memory");
}
__device__ __forceinline__ void wait_async0() {
    asm volatile("s_wait_asynccnt 0x0" ::: "memory");
}

// A fragment (16x32 bf16) from LDS, row-major rows of `stride` halves.
// lane L: g=L>>4, m=L&15 -> row m; halves 0..7 = k in [kb+8g,+8),
// halves 8..15 = k in [kb+16+8g,+8)  (ISA 16-bit A layout).
__device__ __forceinline__ v16bf load_a_lds(const unsigned short* base, int stride, int kb, int lane) {
    int g = lane >> 4;
    int m = lane & 15;
    const unsigned short* p = base + m * stride + kb + 8 * g;
    FragBF r;
    r.u[0] = *(const uint4*)(p);
    r.u[1] = *(const uint4*)(p + 16);
    return r.v;
}

// B fragment (32x16 bf16) from LDS panel: lane L holds column n=nb+(L&15),
// halves 0..7 = k in [kb+8g,+8), halves 8..15 = k in [kb+16+8g,+8) (g=L>>4).
__device__ __forceinline__ v16bf load_b_lds(const unsigned short* base, int stride, int kb, int nb, int lane) {
    const __bf16* p = (const __bf16*)(base + (kb + 8 * (lane >> 4)) * stride + nb + (lane & 15));
    FragBF r;
#pragma unroll
    for (int s = 0; s < 8; ++s) {
        r.e[s]     = p[s * stride];
        r.e[s + 8] = p[(16 + s) * stride];
    }
    return r.v;
}

// B fragment (32x16 bf16) straight from global bf16 (L2-resident weights)
__device__ __forceinline__ v16bf load_b_global_bf16(const unsigned short* __restrict__ B, int ldb,
                                                    int kb, int nb, int lane) {
    const __bf16* p = (const __bf16*)(B + (size_t)(kb + 8 * (lane >> 4)) * ldb + nb + (lane & 15));
    FragBF r;
#pragma unroll
    for (int s = 0; s < 8; ++s) {
        r.e[s]     = p[(size_t)s * ldb];
        r.e[s + 8] = p[(size_t)(16 + s) * ldb];
    }
    return r.v;
}

// B fragment (32x16) from global fp32 with on-the-fly convert (fallback path)
__device__ __forceinline__ v16bf load_b_global_f32(const float* __restrict__ B, int ldb,
                                                   int kb, int nb, int lane) {
    const float* p = B + (size_t)(kb + 8 * (lane >> 4)) * ldb + nb + (lane & 15);
    FragBF r;
#pragma unroll
    for (int s = 0; s < 8; ++s) {
        r.e[s]     = (__bf16)p[(size_t)s * ldb];
        r.e[s + 8] = (__bf16)p[(size_t)(16 + s) * ldb];
    }
    return r.v;
}

// ---------------- kernel 0: zero output + counters ----------------
__global__ void zero_kernel(float* __restrict__ out, int n, int* __restrict__ cnt) {
    int i = blockIdx.x * blockDim.x + threadIdx.x;
    if (i < n) out[i] = 0.0f;
    if (blockIdx.x == 0 && threadIdx.x < 9) cnt[threadIdx.x] = 0;  // cnt[8] + tot[1]
}

// ---------------- kernel 0b: fp32 -> bf16 weight conversion ----------------
__global__ void convert_bf16_kernel(const float* __restrict__ src,
                                    unsigned short* __restrict__ dst, int n) {
    int i = (blockIdx.x * blockDim.x + threadIdx.x) * 8;
    if (i + 7 < n) {
        float4 a = *(const float4*)(src + i);
        float4 b = *(const float4*)(src + i + 4);
        dst[i + 0] = bfbits(a.x); dst[i + 1] = bfbits(a.y);
        dst[i + 2] = bfbits(a.z); dst[i + 3] = bfbits(a.w);
        dst[i + 4] = bfbits(b.x); dst[i + 5] = bfbits(b.y);
        dst[i + 6] = bfbits(b.z); dst[i + 7] = bfbits(b.w);
    }
}

// ---------------- kernel 1: router (one wave32 per token) ----------------
__global__ __launch_bounds__(256) void router_kernel(const float* __restrict__ x,
                                                     const float* __restrict__ Wr,
                                                     int* __restrict__ cnt,
                                                     int* __restrict__ sTok,
                                                     float* __restrict__ sW) {
    int gw   = (blockIdx.x * blockDim.x + threadIdx.x) >> 5;  // token id
    int lane = threadIdx.x & 31;
    if (gw >= NTOK) return;
    const float* xr = x + (size_t)gw * CDIM;

    float p[NEXP];
#pragma unroll
    for (int e = 0; e < NEXP; ++e) p[e] = 0.0f;

    for (int c = lane; c < CDIM; c += 32) {
        float xv = xr[c];
#pragma unroll
        for (int e = 0; e < NEXP; ++e) p[e] += xv * Wr[e * CDIM + c];
    }
#pragma unroll
    for (int off = 16; off > 0; off >>= 1) {
#pragma unroll
        for (int e = 0; e < NEXP; ++e) p[e] += __shfl_xor(p[e], off, 32);
    }

    if (lane == 0) {
        float m = p[0];
#pragma unroll
        for (int e = 1; e < NEXP; ++e) m = fmaxf(m, p[e]);
        float q[NEXP];
#pragma unroll
        for (int e = 0; e < NEXP; ++e) q[e] = expf(p[e] - m);

        int i0 = 0;
#pragma unroll
        for (int e = 1; e < NEXP; ++e) if (q[e] > q[i0]) i0 = e;
        int i1 = (i0 == 0) ? 1 : 0;
#pragma unroll
        for (int e = 0; e < NEXP; ++e) if (e != i0 && q[e] > q[i1]) i1 = e;

        float s  = q[i0] + q[i1];
        float w0 = q[i0] / s;
        float w1 = q[i1] / s;

        int s0 = atomicAdd(&cnt[i0], 1);
        sTok[i0 * NTOK + s0] = gw;
        sW[i0 * NTOK + s0]   = w0;
        int s1 = atomicAdd(&cnt[i1], 1);
        sTok[i1 * NTOK + s1] = gw;
        sW[i1 * NTOK + s1]   = w1;
    }
}

// ---------------- kernel 2: build tile worklist ----------------
__global__ void tile_build_kernel(const int* __restrict__ cnt, int* __restrict__ tot,
                                  int* __restrict__ tileE, int* __restrict__ tileR) {
    if (blockIdx.x == 0 && threadIdx.x == 0) {
        int t = 0;
        for (int e = 0; e < NEXP; ++e) {
            int ntile = (cnt[e] + TM - 1) / TM;
            for (int i = 0; i < ntile; ++i) {
                if (t < TILE_TAB) { tileE[t] = e; tileR[t] = i * TM; }
                ++t;
            }
        }
        tot[0] = (t > TILE_TAB) ? TILE_TAB : t;
    }
}

// =====================================================================
// kernel 3a: grouped FFN, bf16 weights; async up-panel prefetch pipeline:
//   wait(Pan hc) -> upGEMM(Pan) -> barrier -> issue Pan(hc+1) -> downGEMM(global B)
// =====================================================================
__global__ __launch_bounds__(512) void ffn_kernel_bf16(const float* __restrict__ x,
                                                       const unsigned short* __restrict__ W1b,
                                                       const unsigned short* __restrict__ W2b,
                                                       float* __restrict__ out,
                                                       const int* __restrict__ cnt,
                                                       const int* __restrict__ tot,
                                                       const int* __restrict__ tileE,
                                                       const int* __restrict__ tileR,
                                                       const int* __restrict__ sTok,
                                                       const float* __restrict__ sW) {
    __shared__ __align__(16) unsigned short Xs[TM][CS];        // 64x1024 bf16 tokens (130KB)
    __shared__ __align__(16) unsigned short Pan[1024 * PS_UP]; // W1 panel [1024][72] (144KB)
    __shared__ __align__(16) unsigned short Hs[TM][HSF];       // 64x64 bf16 gelu(up) (9KB)
    __shared__ int   tokS[TM];
    __shared__ float wS[TM];

    int bid = blockIdx.x;
    if (bid >= tot[0]) return;

    const int e  = tileE[bid];
    const int rs = tileR[bid];
    const int ce = cnt[e];
    const unsigned short* W1e = W1b + (size_t)e * CDIM * HDIM;  // [C][H] bf16
    const unsigned short* W2e = W2b + (size_t)e * HDIM * CDIM;  // [H][C] bf16

    const int tid  = threadIdx.x;
    const int lane = tid & 31;
    const int w    = tid >> 5;      // wave id 0..15

    // per-thread prefetch bases: thread copies rows k0, k0+64, ... seg s0 (fixed)
    const int k0 = tid >> 3;
    const int s0 = (tid & 7) * 8;
    const unsigned short* pfG0 = W1e + (size_t)k0 * HDIM + s0;  // + hBase per chunk
    unsigned short*       pfL0 = &Pan[k0 * PS_UP + s0];

    if (tid < TM) {
        int idx    = rs + tid;
        bool valid = idx < ce;
        tokS[tid] = valid ? sTok[e * NTOK + idx] : 0;
        wS[tid]   = valid ? sW[e * NTOK + idx] : 0.0f;
    }
    __syncthreads();

    // ---- prefetch chunk-0 up panel (overlaps with the X gather below) ----
    {
        const unsigned short* g = pfG0;
        unsigned short*       l = pfL0;
#pragma unroll
        for (int j = 0; j < 16; ++j) {
            async_copy16(g, l);
            g += 64 * HDIM;
            l += 64 * PS_UP;
        }
    }

    // ---- gather X rows -> bf16 LDS (8 threads per row, 128 cols each) ----
    {
        int r  = tid >> 3;
        int c0 = (tid & 7) * 128;
        bool valid = (wS[r] != 0.0f);
        const float* xr = x + (size_t)tokS[r] * CDIM;
        for (int c = c0; c < c0 + 128; c += 4) {
            float4 f;
            if (valid) f = *(const float4*)(xr + c);
            else       f = make_float4(0.f, 0.f, 0.f, 0.f);
            Xs[r][c + 0] = bfbits(f.x);
            Xs[r][c + 1] = bfbits(f.y);
            Xs[r][c + 2] = bfbits(f.z);
            Xs[r][c + 3] = bfbits(f.w);
        }
    }

    const v8f vzero = {0.f, 0.f, 0.f, 0.f, 0.f, 0.f, 0.f, 0.f};
    v8f acc[16];
#pragma unroll
    for (int i = 0; i < 16; ++i) acc[i] = vzero;

    const int mtU = w & 3;           // up: row tile, col tile = w>>2 (4x4 tiles)
    const int ntU = w >> 2;
    const int rD  = w & 3;           // down: row tile
    const int cb  = (w >> 2) * 256;  // down: col base (16 WMMA col tiles)
    const int g   = lane >> 4;
    const int lm  = lane & 15;
    const unsigned short* xaBase = &Xs[mtU * 16][0];
    const unsigned short* haBase = &Hs[rD * 16][0];

    for (int hc = 0; hc < HDIM / HBF; ++hc) {
        const int hBase = hc * HBF;

        wait_async0();     // this wave's panel transfers done
        __syncthreads();   // panel (and, at hc==0, Xs) visible to all waves

        // ---- up GEMM: h = gelu(X[64,1024] @ Pan) ; one 16x16 tile per wave ----
        {
            v8f c = vzero;
            for (int kk = 0; kk < CDIM / 32; ++kk) {
                v16bf a = load_a_lds(xaBase, CS, kk * 32, lane);
                v16bf b = load_b_lds(&Pan[0], PS_UP, kk * 32, ntU * 16, lane);
                c = __builtin_amdgcn_wmma_f32_16x16x32_bf16(false, a, false, b,
                                                            (short)0, c, false, false);
            }
#pragma unroll
            for (int v = 0; v < 8; ++v)
                Hs[mtU * 16 + g * 8 + v][ntU * 16 + lm] = bfbits(gelu_exact(c[v]));
        }
        __syncthreads();   // all panel reads + Hs writes complete

        // ---- prefetch next chunk's up panel; overlaps with down GEMM ----
        if (hc + 1 < HDIM / HBF) {
            const unsigned short* gp = pfG0 + (hc + 1) * HBF;
            unsigned short*       lp = pfL0;
#pragma unroll
            for (int j = 0; j < 16; ++j) {
                async_copy16(gp, lp);
                gp += 64 * HDIM;
                lp += 64 * PS_UP;
            }
        }

        // ---- down GEMM: acc += h[64,64] @ W2[hBase:+64, :], B from L2 bf16 ----
        {
            const unsigned short* W2c = W2e + (size_t)hBase * CDIM;
            v16bf af[2];
#pragma unroll
            for (int kk = 0; kk < 2; ++kk)
                af[kk] = load_a_lds(haBase, HSF, kk * 32, lane);
#pragma unroll 2
            for (int ct = 0; ct < 16; ++ct) {
#pragma unroll
                for (int kk = 0; kk < 2; ++kk) {
                    v16bf b = load_b_global_bf16(W2c, CDIM, kk * 32, cb + ct * 16, lane);
                    acc[ct] = __builtin_amdgcn_wmma_f32_16x16x32_bf16(false, af[kk], false, b,
                                                                      (short)0, acc[ct], false, false);
                }
            }
        }
        // loop-top wait_async + barrier provide the next-iteration ordering
    }

    // ---- weighted scatter-add into out ----
#pragma unroll 2
    for (int ct = 0; ct < 16; ++ct) {
#pragma unroll
        for (int v = 0; v < 8; ++v) {
            int row   = rD * 16 + g * 8 + v;
            float wgt = wS[row];
            if (wgt != 0.0f) {
                int tok = tokS[row];
                atomicAdd(&out[(size_t)tok * CDIM + cb + ct * 16 + lm], wgt * acc[ct][v]);
            }
        }
    }
}

// =====================================================================
// kernel 3b: fallback — fp32 weights, on-the-fly bf16 convert
// =====================================================================
__global__ __launch_bounds__(512) void ffn_kernel_f32(const float* __restrict__ x,
                                                      const float* __restrict__ W1,
                                                      const float* __restrict__ W2,
                                                      float* __restrict__ out,
                                                      const int* __restrict__ cnt,
                                                      const int* __restrict__ tot,
                                                      const int* __restrict__ tileE,
                                                      const int* __restrict__ tileR,
                                                      const int* __restrict__ sTok,
                                                      const float* __restrict__ sW) {
    __shared__ __align__(16) unsigned short Xs[TM][CS];
    __shared__ __align__(16) unsigned short Hs[TM][HS32];
    __shared__ int   tokS[TM];
    __shared__ float wS[TM];

    int bid = blockIdx.x;
    if (bid >= tot[0]) return;

    const int e  = tileE[bid];
    const int rs = tileR[bid];
    const int ce = cnt[e];
    const float* W1e = W1 + (size_t)e * CDIM * HDIM;
    const float* W2e = W2 + (size_t)e * HDIM * CDIM;

    const int tid  = threadIdx.x;
    const int lane = tid & 31;
    const int w    = tid >> 5;

    if (tid < TM) {
        int idx    = rs + tid;
        bool valid = idx < ce;
        tokS[tid] = valid ? sTok[e * NTOK + idx] : 0;
        wS[tid]   = valid ? sW[e * NTOK + idx] : 0.0f;
    }
    __syncthreads();

    {
        int r  = tid >> 3;
        int c0 = (tid & 7) * 128;
        bool valid = (wS[r] != 0.0f);
        const float* xr = x + (size_t)tokS[r] * CDIM;
        for (int c = c0; c < c0 + 128; c += 4) {
            float4 f;
            if (valid) f = *(const float4*)(xr + c);
            else       f = make_float4(0.f, 0.f, 0.f, 0.f);
            Xs[r][c + 0] = bfbits(f.x);
            Xs[r][c + 1] = bfbits(f.y);
            Xs[r][c + 2] = bfbits(f.z);
            Xs[r][c + 3] = bfbits(f.w);
        }
    }
    __syncthreads();

    const v8f vzero = {0.f, 0.f, 0.f, 0.f, 0.f, 0.f, 0.f, 0.f};
    v8f acc[16];
#pragma unroll
    for (int i = 0; i < 16; ++i) acc[i] = vzero;

    const int mtU = w & 3;
    const int ntB = (w >> 2) * 2;
    const int rD  = w & 3;
    const int cb  = (w >> 2) * 256;
    const int g   = lane >> 4;
    const int lm  = lane & 15;

    for (int hc = 0; hc < HDIM / HB32; ++hc) {
        const int hBase = hc * HB32;
#pragma unroll
        for (int i = 0; i < 2; ++i) {
            int nt = ntB + i;
            v8f c = vzero;
            for (int kk = 0; kk < CDIM / 32; ++kk) {
                v16bf a = load_a_lds(&Xs[mtU * 16][0], CS, kk * 32, lane);
                v16bf b = load_b_global_f32(W1e, HDIM, kk * 32, hBase + nt * 16, lane);
                c = __builtin_amdgcn_wmma_f32_16x16x32_bf16(false, a, false, b,
                                                            (short)0, c, false, false);
            }
#pragma unroll
            for (int v = 0; v < 8; ++v)
                Hs[mtU * 16 + g * 8 + v][nt * 16 + lm] = bfbits(gelu_exact(c[v]));
        }
        __syncthreads();

        v16bf af[4];
#pragma unroll
        for (int kk = 0; kk < 4; ++kk)
            af[kk] = load_a_lds(&Hs[rD * 16][0], HS32, kk * 32, lane);

#pragma unroll 2
        for (int ct = 0; ct < 16; ++ct) {
#pragma unroll
            for (int kk = 0; kk < 4; ++kk) {
                v16bf b = load_b_global_f32(W2e, CDIM, hBase + kk * 32, cb + ct * 16, lane);
                acc[ct] = __builtin_amdgcn_wmma_f32_16x16x32_bf16(false, af[kk], false, b,
                                                                  (short)0, acc[ct], false, false);
            }
        }
        __syncthreads();
    }

#pragma unroll 2
    for (int ct = 0; ct < 16; ++ct) {
#pragma unroll
        for (int v = 0; v < 8; ++v) {
            int row   = rD * 16 + g * 8 + v;
            float wgt = wS[row];
            if (wgt != 0.0f) {
                int tok = tokS[row];
                atomicAdd(&out[(size_t)tok * CDIM + cb + ct * 16 + lm], wgt * acc[ct][v]);
            }
        }
    }
}

// ---------------- host launcher ----------------
extern "C" void kernel_launch(void* const* d_in, const int* in_sizes, int n_in,
                              void* d_out, int out_size, void* d_ws, size_t ws_size,
                              hipStream_t stream) {
    const float* x  = (const float*)d_in[0];
    const float* Wr = (const float*)d_in[1];
    const float* W1 = (const float*)d_in[2];
    const float* W2 = (const float*)d_in[3];
    float* out = (float*)d_out;

    // workspace layout: tables, then (optional) bf16 weight copies
    int*   cnt   = (int*)d_ws;             // [8]
    int*   tot   = cnt + 8;                // [1] (pad to 16)
    int*   tileE = cnt + 16;               // [TILE_TAB]
    int*   tileR = tileE + TILE_TAB;       // [TILE_TAB]
    int*   sTok  = tileR + TILE_TAB;       // [NEXP*NTOK]
    float* sW    = (float*)(sTok + NEXP * NTOK);  // [NEXP*NTOK]

    size_t tabBytes = (size_t)(16 + 2 * TILE_TAB + 2 * NEXP * NTOK) * 4;
    size_t tabAligned = (tabBytes + 255) & ~(size_t)255;
    size_t wElems = (size_t)NEXP * CDIM * HDIM;            // per weight tensor
    size_t need   = tabAligned + wElems * 2 /*tensors*/ * 2 /*bytes*/;
    bool useBf16  = ws_size >= need;

    zero_kernel<<<(out_size + 255) / 256, 256, 0, stream>>>(out, out_size, cnt);
    router_kernel<<<NTOK / 8, 256, 0, stream>>>(x, Wr, cnt, sTok, sW);
    tile_build_kernel<<<1, 1, 0, stream>>>(cnt, tot, tileE, tileR);

    if (useBf16) {
        unsigned short* W1b = (unsigned short*)((char*)d_ws + tabAligned);
        unsigned short* W2b = W1b + wElems;
        int n = (int)wElems;
        convert_bf16_kernel<<<(n / 8 + 255) / 256, 256, 0, stream>>>(W1, W1b, n);
        convert_bf16_kernel<<<(n / 8 + 255) / 256, 256, 0, stream>>>(W2, W2b, n);
        ffn_kernel_bf16<<<MAXT_GRID, 512, 0, stream>>>(x, W1b, W2b, out, cnt, tot,
                                                       tileE, tileR, sTok, sW);
    } else {
        ffn_kernel_f32<<<MAXT_GRID, 512, 0, stream>>>(x, W1, W2, out, cnt, tot,
                                                      tileE, tileR, sTok, sW);
    }
}
